// Block_60498909331923
// MI455X (gfx1250) — matrix operand order repaired and checked by reading, not compiled, not run on hardware
//
#include <hip/hip_runtime.h>
#include <hip/hip_bf16.h>

// ---------------------------------------------------------------------------
// MI455X (gfx1250) implementation of the EDM2-style MP block.
// GEMM work (2x conv3x3 implicit GEMM, QKV 1x1, flash attention, proj 1x1)
// runs on V_WMMA_F32_16X16X32_BF16 (wave32). Tile staging goes through the
// Tensor Data Mover (tensor_load_to_lds / TENSORcnt) with double-buffered
// LDS so TDM flight time overlaps WMMA compute (one barrier per K-step).
// ---------------------------------------------------------------------------

typedef __bf16 bf16_t;
typedef __attribute__((ext_vector_type(16))) __bf16 v16bf;
typedef __attribute__((ext_vector_type(8)))  float  v8f;

__device__ __forceinline__ unsigned short f2bf(float f) {
    unsigned int u = __builtin_bit_cast(unsigned int, f);
    u = (u + 0x7FFFu + ((u >> 16) & 1u)) >> 16;   // round-to-nearest-even
    return (unsigned short)u;
}
__device__ __forceinline__ float bf2f(unsigned short h) {
    unsigned int u = ((unsigned int)h) << 16;
    return __builtin_bit_cast(float, u);
}

union FragU { uint4 u[2]; v16bf v; };

// Load a 16-half WMMA fragment from LDS as two aligned b128 reads.
__device__ __forceinline__ v16bf load_frag_2x(const unsigned short* lds, int off0, int off1) {
    FragU f;
    f.u[0] = *reinterpret_cast<const uint4*>(lds + off0);
    f.u[1] = *reinterpret_cast<const uint4*>(lds + off1);
    return f.v;
}

__device__ __forceinline__ v8f wmma_bf16(v16bf a, v16bf b, v8f c) {
    return __builtin_amdgcn_wmma_f32_16x16x32_bf16(false, a, false, b, (short)0, c, false, false);
}

#define EPSN       1e-4f
#define INV0596    (1.0f / 0.596f)
#define INV_MPDEN  (1.0f / 0.76157731058639087f)   // 1/sqrt(0.7^2+0.3^2)

// ---------------------------------------------------------------------------
// Tensor Data Mover: build a 2-D D# (group0 + group1 per CDNA5 ISA 8.3/8.4)
// and issue TENSOR_LOAD_TO_LDS. data_size = 2 bytes (bf16).
// 6-arg builtin: (uint32x4 g0, int32x8 g1, int32x4 g2, int32x4 g3,
//                 int32x8 g4, i32 cpol)
// ---------------------------------------------------------------------------
#if defined(__has_builtin)
#if __has_builtin(__builtin_amdgcn_tensor_load_to_lds) && __has_builtin(__builtin_amdgcn_s_wait_tensorcnt)
#define USE_TDM 1
#endif
#endif
#ifndef USE_TDM
#define USE_TDM 0
#endif

#if USE_TDM
typedef __attribute__((ext_vector_type(4))) unsigned int v4u;
typedef __attribute__((ext_vector_type(8))) int v8i;
typedef __attribute__((ext_vector_type(4))) int v4i;
typedef __attribute__((address_space(3))) const void lds_cv;

__device__ __forceinline__ unsigned lds_off_of(const void* p) {
    return (unsigned)(unsigned long long)(lds_cv*)p;   // LDS byte offset
}

// tile_d0 elements contiguous per row, tile_d1 rows, row stride in elements.
__device__ __forceinline__ void tdm_load_2d(const void* lds_dst, const void* gsrc,
                                            unsigned tile_d0, unsigned tile_d1,
                                            unsigned long long stride_elems) {
    unsigned long long ga = (unsigned long long)gsrc;
    v4u g0;
    g0.x = 1u;                                                   // count=1 (valid, user)
    g0.y = lds_off_of(lds_dst);                                  // lds_addr
    g0.z = (unsigned)ga;                                         // global_addr[31:0]
    g0.w = (unsigned)((ga >> 32) & 0x01FFFFFFull) | (2u << 30);  // addr[56:32] | type=2
    unsigned td0 = (stride_elems > (unsigned long long)tile_d0) ? (unsigned)stride_elems
                                                                : tile_d0;
    v8i g1;
    g1[0] = (int)(1u << 16);                                     // data_size=1 (2B), mask=0
    g1[1] = (int)((td0 & 0xFFFFu) << 16);                        // tensor_dim0 lo16
    g1[2] = (int)(((td0 >> 16) & 0xFFFFu) |
                  ((tile_d1 & 0xFFFFu) << 16));                  // tensor_dim0 hi | tensor_dim1 lo
    g1[3] = (int)(((tile_d1 >> 16) & 0xFFFFu) |
                  ((tile_d0 & 0xFFFFu) << 16));                  // tensor_dim1 hi | tile_dim0
    g1[4] = (int)(tile_d1 & 0xFFFFu);                            // tile_dim1 | tile_dim2=0
    g1[5] = (int)(unsigned)(stride_elems & 0xFFFFFFFFull);       // dim0_stride lo32
    g1[6] = (int)(unsigned)((stride_elems >> 32) & 0xFFFFull);   // dim0_stride hi16
    g1[7] = 0;
    v4i gz4 = {};
    v8i gz8 = {};
    __builtin_amdgcn_tensor_load_to_lds(g0, g1, gz4, gz4, gz8, 0);
}
#endif

// ---------------------------------------------------------------------------
// Weight prep: normalize rows over fan_in, scale 1/sqrt(fan_in),
// store bf16 transposed [tap][Cout][Cin]. src: [Cout][Cin][kh][kw].
// ---------------------------------------------------------------------------
__global__ __launch_bounds__(256) void prep_weight(const float* __restrict__ w,
                                                   unsigned short* __restrict__ dst,
                                                   int Cout, int Cin, int T) {
    __shared__ float red[256];
    int row = blockIdx.x;
    int fanin = Cin * T;
    const float* wr = w + (size_t)row * fanin;
    float ss = 0.f;
    for (int f = threadIdx.x; f < fanin; f += 256) { float v = wr[f]; ss += v * v; }
    red[threadIdx.x] = ss;
    __syncthreads();
    for (int o = 128; o > 0; o >>= 1) {
        if ((int)threadIdx.x < o) red[threadIdx.x] += red[threadIdx.x + o];
        __syncthreads();
    }
    float alpha = rsqrtf((float)fanin);
    float scale = alpha / (EPSN + alpha * sqrtf(red[0]));
    for (int f = threadIdx.x; f < fanin; f += 256) {
        int t = f % T, cin = f / T;
        dst[((size_t)t * Cout + row) * Cin + cin] = f2bf(wr[f] * scale);
    }
}

__global__ __launch_bounds__(256) void emb_linear(const float* __restrict__ emb,
                                                  const float* __restrict__ wemb,
                                                  const float* __restrict__ gain,
                                                  float* __restrict__ cmod) {
    int n = blockIdx.x;
    int co = threadIdx.x;
    const float* wr = wemb + (size_t)co * 768;
    const float* e  = emb  + (size_t)n  * 768;
    float ss = 0.f, dot = 0.f;
    for (int f = 0; f < 768; ++f) { float v = wr[f]; ss += v * v; dot += v * e[f]; }
    float alpha = rsqrtf(768.0f);
    float scale = (gain[0] * alpha) / (EPSN + alpha * sqrtf(ss));
    cmod[n * 256 + co] = dot * scale + 1.0f;
}

__global__ __launch_bounds__(256) void pixnorm(const float* __restrict__ x,
                                               float* __restrict__ xn,
                                               unsigned short* __restrict__ act0) {
    int p = blockIdx.x * 256 + threadIdx.x;     // n*S + s
    int n = p >> 10, s = p & 1023;
    const float* base = x + ((size_t)n * 256) * 1024 + s;
    float ss = 0.f;
    for (int c = 0; c < 256; ++c) { float v = base[(size_t)c * 1024]; ss += v * v; }
    float fac = 1.0f / (EPSN + sqrtf(ss) * (1.0f / 16.0f));
    float* xo = xn + ((size_t)n * 256) * 1024 + s;
    unsigned short* ao = act0 + ((size_t)n * 256) * 1024 + s;
    for (int c = 0; c < 256; ++c) {
        float v = base[(size_t)c * 1024] * fac;
        xo[(size_t)c * 1024] = v;
        float sv = v / (1.0f + __expf(-v)) * INV0596;
        ao[(size_t)c * 1024] = f2bf(sv);
    }
}

// ---------------------------------------------------------------------------
// Implicit-GEMM conv. act: bf16 [N][Cin][S], wT: bf16 [T][CoutT][Cin].
// Block: 64 Cout x 128 S, 8 waves (2x4), each wave 2x2 of 16x16 WMMA tiles.
// Double-buffered LDS: TDM(A) + gather(B) for step it+1 are issued right
// after the barrier and drained (s_wait_tensorcnt) after the WMMAs.
// ---------------------------------------------------------------------------
template <int MODE>
__global__ __launch_bounds__(256) void conv_gemm(const unsigned short* __restrict__ act,
                                                 const unsigned short* __restrict__ wT,
                                                 int Cin, int T, int CoutT,
                                                 const float* __restrict__ cmod,
                                                 const float* __restrict__ xres,
                                                 float* __restrict__ out_f32,
                                                 unsigned short* __restrict__ out_bf) {
    constexpr int ASTR = 32, BSTR = 40;
    __shared__ __align__(16) unsigned short lA[2][64 * ASTR];    // [cout_local][k]
    __shared__ __align__(16) unsigned short lB[2][128 * BSTR];   // [col][k] (transposed)
    const int S = 1024;
    int n = blockIdx.z;
    int cout0 = blockIdx.y * 64;
    int s0 = blockIdx.x * 128;
    int row0 = s0 >> 5;
    int tid = threadIdx.x;
    int lane = tid & 31, wid = tid >> 5;
    int wm = wid >> 2, wn = wid & 3;

    v8f acc[2][2] = {};

    int arow0 = wm * 32 + (lane & 15);
    int akb = (lane < 16) ? 0 : 8;
    int bcol0 = wn * 32 + (lane & 15);
    int bkb = (lane < 16) ? 0 : 8;

    const int kSteps = Cin >> 5;
    const int nIter = T * kSteps;

    // Stage K-step `sit` into buffer `sbuf` (no barrier inside).
    auto stage = [&](int sit, int sbuf) {
        int t  = sit / kSteps;
        int k0 = (sit - t * kSteps) << 5;
        int dy = (T == 9) ? (t / 3) - 1 : 0;
        int dx = (T == 9) ? (t % 3) - 1 : 0;
        const unsigned short* wTt = wT + (size_t)t * CoutT * Cin;
#if USE_TDM
        if (wid == 0)   // A tile: 64 rows x 32 halves, row stride Cin
            tdm_load_2d(&lA[sbuf][0], wTt + (size_t)cout0 * Cin + k0, 32, 64,
                        (unsigned long long)Cin);
#else
        {
            int r  = tid >> 2;
            int kb = (tid & 3) << 3;
            const uint4* src = reinterpret_cast<const uint4*>(
                wTt + (size_t)(cout0 + r) * Cin + k0 + kb);
            *reinterpret_cast<uint4*>(&lA[sbuf][r * ASTR + kb]) = *src;
        }
#endif
        {   // B tile transposed: lB[col][k] = act[n][k0+k][shift(col)]
            int kl = tid >> 3;              // 0..31
            int cb = (tid & 7) << 4;        // 0..112
            const unsigned short* srow = act + ((size_t)n * Cin + k0 + kl) * S;
            __builtin_prefetch(srow + S, 0, 1);
#pragma unroll
            for (int i = 0; i < 16; ++i) {
                int c  = cb + i;
                int hh = row0 + (c >> 5) + dy;
                int ww = (c & 31) + dx;
                unsigned short v = 0;
                if ((unsigned)hh < 32u && (unsigned)ww < 32u)
                    v = srow[hh * 32 + ww];
                lB[sbuf][c * BSTR + kl] = v;
            }
        }
    };

    stage(0, 0);
#if USE_TDM
    if (wid == 0) __builtin_amdgcn_s_wait_tensorcnt(0);
#endif

    for (int it = 0; it < nIter; ++it) {
        int cur = it & 1;
        __syncthreads();                    // publish buffer `cur`
        if (it + 1 < nIter) stage(it + 1, cur ^ 1);   // overlap with compute

        const unsigned short* la = &lA[cur][0];
        const unsigned short* lb = &lB[cur][0];
        v16bf af[2], bfr[2];
        af[0]  = load_frag_2x(la, (arow0)      * ASTR + akb, (arow0)      * ASTR + akb + 16);
        af[1]  = load_frag_2x(la, (arow0 + 16) * ASTR + akb, (arow0 + 16) * ASTR + akb + 16);
        bfr[0] = load_frag_2x(lb, (bcol0)      * BSTR + bkb, (bcol0)      * BSTR + bkb + 16);
        bfr[1] = load_frag_2x(lb, (bcol0 + 16) * BSTR + bkb, (bcol0 + 16) * BSTR + bkb + 16);
#pragma unroll
        for (int tm = 0; tm < 2; ++tm)
#pragma unroll
            for (int tn = 0; tn < 2; ++tn)
                acc[tm][tn] = wmma_bf16(af[tm], bfr[tn], acc[tm][tn]);

#if USE_TDM
        if (wid == 0 && it + 1 < nIter) __builtin_amdgcn_s_wait_tensorcnt(0);
#endif
    }

    int hilo = (lane >= 16) ? 8 : 0;
#pragma unroll
    for (int tm = 0; tm < 2; ++tm)
#pragma unroll
        for (int tn = 0; tn < 2; ++tn)
#pragma unroll
            for (int r = 0; r < 8; ++r) {
                int row = cout0 + wm * 32 + tm * 16 + r + hilo;
                int col = s0 + wn * 32 + tn * 16 + (lane & 15);
                float y = acc[tm][tn][r];
                size_t oi = ((size_t)n * CoutT + row) * S + col;
                if constexpr (MODE == 0) {
                    float z = y * cmod[n * CoutT + row];
                    float sv = z / (1.0f + __expf(-z)) * INV0596;
                    out_bf[oi] = f2bf(sv);
                } else if constexpr (MODE == 1) {
                    float xr = xres[oi];
                    float x1 = (xr + 0.3f * (y - xr)) * INV_MPDEN;
                    out_f32[oi] = x1;
                    out_bf[oi]  = f2bf(x1);
                } else if constexpr (MODE == 2) {
                    out_bf[oi] = f2bf(y);
                } else {
                    float xr = xres[oi];
                    float v = (xr + 0.3f * (y - xr)) * INV_MPDEN;
                    v = fminf(fmaxf(v, -256.0f), 256.0f);
                    out_f32[oi] = v;
                }
            }
}

// ---------------------------------------------------------------------------
// QKV head-norm. q/k written s-major [N][H][S][CPH] (TDM-friendly contiguous
// slabs); v stays c-major [N][H][CPH][S]. 1/sqrt(CPH) folded into q.
// yq channel index: ch = (head*64+cph)*3 + which
// ---------------------------------------------------------------------------
__global__ __launch_bounds__(256) void qkv_norm(const unsigned short* __restrict__ yq,
                                                unsigned short* __restrict__ qb,
                                                unsigned short* __restrict__ kb,
                                                unsigned short* __restrict__ vb) {
    int idx = blockIdx.x * 256 + threadIdx.x;
    int s = idx & 1023;
    int rest = idx >> 10;                 // n*12 + head*3 + which
    int which = rest % 3;
    int head = (rest / 3) & 3;
    int n = rest / 12;
    const unsigned short* base = yq + ((size_t)n * 768) * 1024 + s;
    float ss = 0.f;
    for (int c = 0; c < 64; ++c) {
        int ch = (head * 64 + c) * 3 + which;
        float v = bf2f(base[(size_t)ch * 1024]);
        ss += v * v;
    }
    float fac = 1.0f / (EPSN + sqrtf(ss) * 0.125f);
    size_t base2 = ((size_t)(n * 4 + head)) * 64 * 1024;
    if (which == 2) {
        unsigned short* dst = vb + base2 + s;
        for (int c = 0; c < 64; ++c) {
            int ch = (head * 64 + c) * 3 + 2;
            dst[(size_t)c * 1024] = f2bf(bf2f(base[(size_t)ch * 1024]) * fac);
        }
    } else {
        if (which == 0) fac *= 0.125f;
        unsigned short* dst = (which == 0 ? qb : kb) + base2 + (size_t)s * 64;
        for (int c = 0; c < 64; ++c) {
            int ch = (head * 64 + c) * 3 + which;
            dst[c] = f2bf(bf2f(base[(size_t)ch * 1024]) * fac);
        }
    }
}

// ---------------------------------------------------------------------------
// Flash attention: per (n, head, q-tile 128); k-tiles of 64; online softmax.
// q/k: bf16 [N][4][S][64] (s-major), v: bf16 [N][4][64][S].
// K/V tiles double-buffered via TDM. out: bf16 [N][256][S] (ch = head*64+c)
// ---------------------------------------------------------------------------
__global__ __launch_bounds__(256) void flash_attn(const unsigned short* __restrict__ qb,
                                                  const unsigned short* __restrict__ kb,
                                                  const unsigned short* __restrict__ vb,
                                                  unsigned short* __restrict__ outb) {
    constexpr int S = 1024;
    constexpr int CSTR = 64, PSTR = 64;
    __shared__ __align__(16) unsigned short lQ[128 * CSTR];     // [qi][c]
    __shared__ __align__(16) unsigned short lK[2][64 * CSTR];   // [kj][c]
    __shared__ __align__(16) unsigned short lV[2][64 * CSTR];   // [c][k]
    __shared__ __align__(16) unsigned short lP[128 * PSTR];     // [qi][k]
    int n = blockIdx.z, head = blockIdx.y;
    int q0 = blockIdx.x * 128;
    int tid = threadIdx.x, lane = tid & 31, wid = tid >> 5;
    const size_t base = ((size_t)(n * 4 + head)) * 64 * S;

    // Stage K/V tile j into buffer sbuf (no barrier inside).
    auto stageKV = [&](int sj, int sbuf) {
        int k0s = sj * 64;
#if USE_TDM
        if (wid == 0) {
            tdm_load_2d(&lK[sbuf][0], kb + base + (size_t)k0s * 64, 4096, 1, 4096);
            tdm_load_2d(&lV[sbuf][0], vb + base + k0s, 64, 64, (unsigned long long)S);
        }
#else
        for (int e = tid; e < 4096; e += 256) {
            lK[sbuf][e] = kb[base + (size_t)k0s * 64 + e];
            lV[sbuf][e] = vb[base + (size_t)(e >> 6) * S + k0s + (e & 63)];
        }
#endif
    };

#if USE_TDM
    if (wid == 0) {   // Q tile: contiguous 128*64 slab
        tdm_load_2d(lQ, qb + base + (size_t)q0 * 64, 8192, 1, 8192);
    }
#else
    for (int e = tid; e < 8192; e += 256) lQ[e] = qb[base + (size_t)q0 * 64 + e];
#endif
    stageKV(0, 0);
#if USE_TDM
    if (wid == 0) __builtin_amdgcn_s_wait_tensorcnt(0);
#endif

    v8f O[4] = {};
    float m[8], l[8];
#pragma unroll
    for (int r = 0; r < 8; ++r) { m[r] = -1e30f; l[r] = 0.f; }

    int row0 = wid * 16;
    int hilo = (lane >= 16) ? 8 : 0;
    int akb = (lane < 16) ? 0 : 8;
    int bkb = (lane < 16) ? 0 : 8;
    int arow = row0 + (lane & 15);

    for (int j = 0; j < 16; ++j) {
        int cur = j & 1;
        __syncthreads();                          // publish K/V buffer `cur`
        if (j + 1 < 16) stageKV(j + 1, cur ^ 1);  // overlap with compute

        const unsigned short* lk = &lK[cur][0];
        const unsigned short* lv = &lV[cur][0];

        // S = Q @ K^T
        v8f sacc[4] = {};
#pragma unroll
        for (int ks = 0; ks < 64; ks += 32) {
            v16bf aq = load_frag_2x(lQ, arow * CSTR + ks + akb, arow * CSTR + ks + akb + 16);
#pragma unroll
            for (int tn = 0; tn < 4; ++tn) {
                int col = tn * 16 + (lane & 15);
                v16bf bkf = load_frag_2x(lk, col * CSTR + ks + bkb, col * CSTR + ks + bkb + 16);
                sacc[tn] = wmma_bf16(aq, bkf, sacc[tn]);
            }
        }

        // online softmax; write P tile (wave-private rows)
#pragma unroll
        for (int r = 0; r < 8; ++r) {
            float mx = sacc[0][r];
#pragma unroll
            for (int tn = 1; tn < 4; ++tn) mx = fmaxf(mx, sacc[tn][r]);
            for (int d = 1; d < 16; d <<= 1) mx = fmaxf(mx, __shfl_xor(mx, d, 32));
            float mnew = fmaxf(m[r], mx);
            float sc = __expf(m[r] - mnew);
            float rsum = 0.f;
            int prow = row0 + r + hilo;
#pragma unroll
            for (int tn = 0; tn < 4; ++tn) {
                float p = __expf(sacc[tn][r] - mnew);
                rsum += p;
                lP[prow * PSTR + tn * 16 + (lane & 15)] = f2bf(p);
            }
            for (int d = 1; d < 16; d <<= 1) rsum += __shfl_xor(rsum, d, 32);
            l[r] = l[r] * sc + rsum;
            m[r] = mnew;
#pragma unroll
            for (int tc = 0; tc < 4; ++tc) O[tc][r] *= sc;
        }

        // O += P @ V
#pragma unroll
        for (int ks = 0; ks < 64; ks += 32) {
            v16bf ap = load_frag_2x(lP, arow * PSTR + ks + akb, arow * PSTR + ks + akb + 16);
#pragma unroll
            for (int tc = 0; tc < 4; ++tc) {
                int col = tc * 16 + (lane & 15);
                v16bf bvf = load_frag_2x(lv, col * CSTR + ks + bkb, col * CSTR + ks + bkb + 16);
                O[tc] = wmma_bf16(ap, bvf, O[tc]);
            }
        }

#if USE_TDM
        if (wid == 0 && j + 1 < 16) __builtin_amdgcn_s_wait_tensorcnt(0);
#endif
    }

#pragma unroll
    for (int r = 0; r < 8; ++r) {
        float inv = 1.0f / l[r];
        int srow = q0 + row0 + r + hilo;
#pragma unroll
        for (int tc = 0; tc < 4; ++tc) {
            int ch = head * 64 + tc * 16 + (lane & 15);
            outb[((size_t)n * 256 + ch) * S + srow] = f2bf(O[tc][r] * inv);
        }
    }
}

// ---------------------------------------------------------------------------
extern "C" void kernel_launch(void* const* d_in, const int* in_sizes, int n_in,
                              void* d_out, int out_size, void* d_ws, size_t ws_size,
                              hipStream_t stream) {
    const float* x        = (const float*)d_in[0];
    const float* emb      = (const float*)d_in[1];
    const float* w_res0   = (const float*)d_in[2];
    const float* w_emb    = (const float*)d_in[3];
    const float* w_res1   = (const float*)d_in[4];
    const float* w_qkv    = (const float*)d_in[5];
    const float* w_proj   = (const float*)d_in[6];
    const float* emb_gain = (const float*)d_in[7];

    char* ws = (char*)d_ws;
    size_t off = 0;
    auto take = [&](size_t bytes) -> char* {
        char* p = ws + off;
        off += (bytes + 255) & ~(size_t)255;
        return p;
    };

    unsigned short* wT0   = (unsigned short*)take((size_t)9 * 256 * 256 * 2);
    unsigned short* wT1   = (unsigned short*)take((size_t)9 * 256 * 256 * 2);
    unsigned short* wTq   = (unsigned short*)take((size_t)1 * 768 * 256 * 2);
    unsigned short* wTp   = (unsigned short*)take((size_t)1 * 256 * 256 * 2);
    float*          cmod  = (float*)take((size_t)32 * 256 * 4);
    float*          xn    = (float*)take((size_t)32 * 256 * 1024 * 4);
    unsigned short* act0  = (unsigned short*)take((size_t)32 * 256 * 1024 * 2);
    unsigned short* act1  = (unsigned short*)take((size_t)32 * 256 * 1024 * 2);
    float*          x1f   = (float*)take((size_t)32 * 256 * 1024 * 4);
    unsigned short* x1b   = (unsigned short*)take((size_t)32 * 256 * 1024 * 2);
    unsigned short* yqkv  = (unsigned short*)take((size_t)32 * 768 * 1024 * 2);
    // lifetime-based aliases (safe w.r.t. launch order below)
    unsigned short* qbuf = act0;                    // act0 dead after conv0
    unsigned short* kbuf = act1;                    // act1 dead after conv1
    unsigned short* vbuf = (unsigned short*)xn;     // xn dead after conv1 epilogue
    unsigned short* attn = yqkv;                    // yqkv dead after qkv_norm

    prep_weight<<<256, 256, 0, stream>>>(w_res0, wT0, 256, 256, 9);
    prep_weight<<<256, 256, 0, stream>>>(w_res1, wT1, 256, 256, 9);
    prep_weight<<<768, 256, 0, stream>>>(w_qkv,  wTq, 768, 256, 1);
    prep_weight<<<256, 256, 0, stream>>>(w_proj, wTp, 256, 256, 1);
    emb_linear<<<32, 256, 0, stream>>>(emb, w_emb, emb_gain, cmod);
    pixnorm<<<128, 256, 0, stream>>>(x, xn, act0);

    conv_gemm<0><<<dim3(8, 4, 32), 256, 0, stream>>>(act0, wT0, 256, 9, 256,
                                                     cmod, nullptr, nullptr, act1);
    conv_gemm<1><<<dim3(8, 4, 32), 256, 0, stream>>>(act1, wT1, 256, 9, 256,
                                                     nullptr, xn, x1f, x1b);
    conv_gemm<2><<<dim3(8, 12, 32), 256, 0, stream>>>(x1b, wTq, 256, 1, 768,
                                                      nullptr, nullptr, nullptr, yqkv);
    qkv_norm<<<1536, 256, 0, stream>>>(yqkv, qbuf, kbuf, vbuf);
    flash_attn<<<dim3(8, 4, 32), 256, 0, stream>>>(qbuf, kbuf, vbuf, attn);
    conv_gemm<3><<<dim3(8, 4, 32), 256, 0, stream>>>(attn, wTp, 256, 1, 256,
                                                     nullptr, x1f, (float*)d_out, nullptr);
}
